// EdgeConv_50818053046294
// MI455X (gfx1250) — matrix operand order repaired and checked by reading, not compiled, not run on hardware
//
#include <hip/hip_runtime.h>
#include <hip/hip_bf16.h>

typedef __attribute__((ext_vector_type(16))) _Float16 v16h;
typedef __attribute__((ext_vector_type(8)))  _Float16 v8h;
typedef __attribute__((ext_vector_type(8)))  float    v8f;
typedef __attribute__((ext_vector_type(4)))  float    v4f;

#define N_NODES 50000
#define N_EDGES 500000
#define NODE_C  128
#define EDGE_C  64
#define OUT_C   128
#define IN_DIM  320            // 2*NODE_C + EDGE_C
#define WAVES   8
#define BLOCK   (WAVES * 32)
#define NTILES  (N_EDGES / 16) // 31250, exact

__launch_bounds__(BLOCK, 1)
__global__ void edgeconv_mlp_kernel(const float* __restrict__ x,
                                    const long long* __restrict__ ei,
                                    const float* __restrict__ ea,
                                    const float* __restrict__ W1,
                                    const float* __restrict__ b1,
                                    const float* __restrict__ W2,
                                    const float* __restrict__ b2,
                                    float* __restrict__ node_acc,   // [N_NODES*OUT_C], pre-zeroed
                                    float* __restrict__ edge_out,   // [N_EDGES*OUT_C]
                                    float* __restrict__ counts)     // [N_NODES], pre-zeroed
{
    // LDS: transposed f16 weights (B-matrix layout wants K contiguous per column)
    __shared__ _Float16 w1t[OUT_C * IN_DIM];        // 80 KB  [n][k]
    __shared__ _Float16 w2t[OUT_C * OUT_C];         // 32 KB  [n][k]
    __shared__ _Float16 htile[WAVES * 16 * OUT_C];  // 32 KB  per-wave 16x128 h staging

    const int tid = threadIdx.x;

    // Stage W1^T and W2^T as f16 (coalesced read, strided LDS write; one-time cost)
    for (int i = tid; i < IN_DIM * OUT_C; i += BLOCK) {
        int k = i >> 7, n = i & 127;               // W1[k][n]
        w1t[n * IN_DIM + k] = (_Float16)W1[i];
    }
    for (int i = tid; i < OUT_C * OUT_C; i += BLOCK) {
        int k = i >> 7, n = i & 127;               // W2[k][n]
        w2t[n * OUT_C + k] = (_Float16)W2[i];
    }
    __syncthreads();

    const int wave   = tid >> 5;
    const int lane   = tid & 31;
    const int n0     = lane & 15;   // column / row-in-tile index
    const int hi     = lane >> 4;   // half-wave select
    const int base8  = hi * 8;      // accumulator row offset (M)
    const int koff16 = hi * 16;     // B-fragment K offset
    _Float16* hw = &htile[wave * 16 * OUT_C];

    const int gwave  = blockIdx.x * WAVES + wave;
    const int nwaves = gridDim.x * WAVES;

    for (int tile = gwave; tile < NTILES; tile += nwaves) {
        const int e0 = tile * 16;
        const int em = e0 + n0;                    // this lane's edge (A row m = n0)
        const long long r = ei[em];
        const long long c = ei[N_EDGES + em];

        // ---- GEMM1: [16 x 320] x [320 x 128], bias-initialized accumulators ----
        v8f acc[8];
        for (int nb = 0; nb < 8; ++nb) {
            float bv = b1[nb * 16 + n0];
            for (int g = 0; g < 8; ++g) acc[nb][g] = bv;
        }

        // A-fragment gather + 8x WMMA for one 32-wide K block starting at
        // concat-feature f0; src points at the owning tensor row.
        // lane<16: K {0..15}; lane>=16: K {16..31}.
        #define KBLOCK(f0, src, NTLOAD)                                              \
        {                                                                            \
            const v4f* pa = (const v4f*)((src) + koff16);                            \
            const v4f* pb = (const v4f*)((src) + koff16 + 8);                        \
            v4f c0, c1, c2, c3;                                                      \
            if (NTLOAD) {                                                            \
                c0 = __builtin_nontemporal_load(pa);                                 \
                c1 = __builtin_nontemporal_load(pa + 1);                             \
                c2 = __builtin_nontemporal_load(pb);                                 \
                c3 = __builtin_nontemporal_load(pb + 1);                             \
            } else {                                                                 \
                c0 = pa[0]; c1 = pa[1]; c2 = pb[0]; c3 = pb[1];                      \
            }                                                                        \
            v16h a;                                                                  \
            for (int i2 = 0; i2 < 4; ++i2) {                                         \
                a[i2]      = (_Float16)c0[i2];                                       \
                a[4 + i2]  = (_Float16)c1[i2];                                       \
                a[8 + i2]  = (_Float16)c2[i2];                                       \
                a[12 + i2] = (_Float16)c3[i2];                                       \
            }                                                                        \
            for (int nb = 0; nb < 8; ++nb) {                                         \
                const v16h* bp =                                                     \
                    (const v16h*)&w1t[(nb * 16 + n0) * IN_DIM + (f0) + koff16];      \
                acc[nb] = __builtin_amdgcn_wmma_f32_16x16x32_f16(                    \
                    false, a, false, *bp, (short)0, acc[nb], false, false);          \
            }                                                                        \
        }

        {
            const float* xi = x + (size_t)r * NODE_C;   // features [0,128)
            for (int kb = 0; kb < 4; ++kb) KBLOCK(kb * 32, xi + kb * 32, 0)
        }
        {
            const float* xj = x + (size_t)c * NODE_C;   // features [128,256)
            for (int kb = 0; kb < 4; ++kb) KBLOCK(128 + kb * 32, xj + kb * 32, 0)
        }
        {
            const float* eav = ea + (size_t)em * EDGE_C; // features [256,320), streamed once
            for (int kb = 0; kb < 2; ++kb) KBLOCK(256 + kb * 32, eav + kb * 32, 1)
        }
        #undef KBLOCK

        // ---- ReLU -> f16 -> LDS h tile (C-layout -> row-major transpose) ----
        for (int nb = 0; nb < 8; ++nb)
            for (int g = 0; g < 8; ++g) {
                float v = acc[nb][g];
                v = v > 0.f ? v : 0.f;
                hw[(g + base8) * OUT_C + nb * 16 + n0] = (_Float16)v;  // row M=g+8*hi
            }
        asm volatile("s_wait_dscnt 0" ::: "memory");  // wave-private LDS transpose fence

        if (hi == 0) {  // one count per edge; overlap with GEMM2 latency
            __hip_atomic_fetch_add(&counts[(int)r], 1.0f, __ATOMIC_RELAXED,
                                   __HIP_MEMORY_SCOPE_AGENT);
        }

        // ---- GEMM2: [16 x 128] x [128 x 128] ----
        v8f acc2[8];
        for (int nb = 0; nb < 8; ++nb) {
            float bv = b2[nb * 16 + n0];
            for (int g = 0; g < 8; ++g) acc2[nb][g] = bv;
        }
        for (int kb = 0; kb < 4; ++kb) {
            const v8h* p0 = (const v8h*)&hw[n0 * OUT_C + kb * 32 + koff16];
            const v8h* p1 = (const v8h*)&hw[n0 * OUT_C + kb * 32 + koff16 + 8];
            v8h lo = *p0, hv = *p1;
            v16h a;
            for (int i2 = 0; i2 < 8; ++i2) { a[i2] = lo[i2]; a[8 + i2] = hv[i2]; }
            for (int nb = 0; nb < 8; ++nb) {
                const v16h* bp = (const v16h*)&w2t[(nb * 16 + n0) * OUT_C + kb * 32 + koff16];
                acc2[nb] = __builtin_amdgcn_wmma_f32_16x16x32_f16(
                    false, a, false, *bp, (short)0, acc2[nb], false, false);
            }
        }
        asm volatile("s_wait_dscnt 0" ::: "memory");  // h reads done before next-iter stores

        // ---- edge_out store (NT: streamed, protect x in L2) + scatter-sum atomics ----
        for (int g = 0; g < 8; ++g) {
            const int row = g + base8;
            const int e   = e0 + row;
            const long long rr = ei[e];
            float* eo = edge_out + (size_t)e  * OUT_C;
            float* na = node_acc + (size_t)rr * OUT_C;
            for (int nb = 0; nb < 8; ++nb) {
                const int col = nb * 16 + n0;
                const float v = acc2[nb][g];
                __builtin_nontemporal_store(v, &eo[col]);
                __hip_atomic_fetch_add(&na[col], v, __ATOMIC_RELAXED,
                                       __HIP_MEMORY_SCOPE_AGENT);
            }
        }
    }
}

__global__ void node_mean_kernel(float* __restrict__ node,
                                 const float* __restrict__ counts)
{
    int i = blockIdx.x * blockDim.x + threadIdx.x;
    if (i < N_NODES * OUT_C) {
        float cnt = counts[i >> 7];
        node[i] = node[i] / fmaxf(cnt, 1.0f);
    }
}

extern "C" void kernel_launch(void* const* d_in, const int* in_sizes, int n_in,
                              void* d_out, int out_size, void* d_ws, size_t ws_size,
                              hipStream_t stream) {
    const float*     x  = (const float*)d_in[0];
    const long long* ei = (const long long*)d_in[1];   // int64 edge_index [2, E]
    const float*     ea = (const float*)d_in[2];
    const float*     W1 = (const float*)d_in[3];
    const float*     b1 = (const float*)d_in[4];
    const float*     W2 = (const float*)d_in[5];
    const float*     b2 = (const float*)d_in[6];

    float* out      = (float*)d_out;
    float* node_acc = out;                               // node_out first in tuple
    float* edge_out = out + (size_t)N_NODES * OUT_C;     // then edge_out
    float* counts   = (float*)d_ws;                      // 200 KB scratch

    hipMemsetAsync(node_acc, 0, (size_t)N_NODES * OUT_C * sizeof(float), stream);
    hipMemsetAsync(counts, 0, (size_t)N_NODES * sizeof(float), stream);

    edgeconv_mlp_kernel<<<256, BLOCK, 0, stream>>>(x, ei, ea, W1, b1, W2, b2,
                                                   node_acc, edge_out, counts);
    node_mean_kernel<<<(N_NODES * OUT_C + 255) / 256, 256, 0, stream>>>(node_acc, counts);
}
